// BiLSTM_CRF_77154792505704
// MI455X (gfx1250) — compile-verified, matrix operand order checked by reading, not run on hardware
//
#include <hip/hip_runtime.h>
#include <hip/hip_bf16.h>
#include <math.h>

// Problem constants (match reference)
#define Vn 50000
#define Bn 64
#define Tn 256
#define Dn 256
#define Hn 256
#define Cn 32
#define G4 1024          // 4*H
#define RT (Bn * Tn)     // 16384 rows of the batch-major flattening (r = b*T + t)

typedef __attribute__((ext_vector_type(16))) _Float16 v16h;
typedef __attribute__((ext_vector_type(8)))  float    v8f;

union FragU { v16h v; unsigned int u[8]; };

// Load a 16x32 f16 fragment for WMMA A (rows=M) or B (rows=N) from a
// row-major [row][stride] f16 array. Per ISA 7.12.2:
//   lane L holds row (L&15); lanes 16-31 hold K+8; VGPR v packs K pairs:
//   k = (v/4)*16 + (L/16)*8 + (v%4)*2, (k,k+1) in one dword.
__device__ __forceinline__ v16h load_frag16x32(const _Float16* __restrict__ base,
                                               int row0, int k0, int stride, int lane) {
  FragU f;
  const _Float16* p = base + (size_t)(row0 + (lane & 15)) * stride + k0 + ((lane >> 4) << 3);
#pragma unroll
  for (int i = 0; i < 4; ++i) {
    f.u[i]     = *(const unsigned int*)(p + 2 * i);        // K = 0..7 (+half*8)
    f.u[i + 4] = *(const unsigned int*)(p + 16 + 2 * i);   // K = 16..23 (+half*8)
  }
  return f.v;
}

__device__ __forceinline__ float sigf(float x) { return 1.0f / (1.0f + __expf(-x)); }

// 16 consecutive floats via 4x float4 (works for LDS and global pointers)
__device__ __forceinline__ void load16f(float* d, const float* s) {
#pragma unroll
  for (int q = 0; q < 4; ++q)
    ((float4*)d)[q] = ((const float4*)s)[q];
}

// ---------------------------------------------------------------------------
// Phase 0: f32 -> f16 weight conversion
// ---------------------------------------------------------------------------
__global__ void cvt_f16_kernel(const float* __restrict__ src, _Float16* __restrict__ dst, int n) {
  int i = blockIdx.x * 256 + threadIdx.x;
  if (i < n) dst[i] = (_Float16)src[i];
}

// ---------------------------------------------------------------------------
// Phase 1: embedding gather + input-gate GEMM (both directions)
//   G[r, n] = emb[sent[r]] . w_ih[n, :] + b_ih[n] + b_hh[n],  r = b*T + t
// Grid: (256 row-blocks of 64, 8 col-blocks of 128, 2 dirs); block 256 thr.
// ---------------------------------------------------------------------------
__global__ void __launch_bounds__(256)
input_gates_kernel(const int* __restrict__ sent, const float* __restrict__ emb,
                   const _Float16* __restrict__ wihF, const _Float16* __restrict__ wihB,
                   const float* __restrict__ bihF, const float* __restrict__ bhhF,
                   const float* __restrict__ bihB, const float* __restrict__ bhhB,
                   float* __restrict__ Gf, float* __restrict__ Gb) {
  __shared__ int toks[64];
  __shared__ _Float16 a_lds[64 * Dn];   // 32 KB
  const int dir = blockIdx.z;
  const int r0  = blockIdx.x * 64;
  const int tid = threadIdx.x;

  if (tid < 64) toks[tid] = sent[r0 + tid];
  __syncthreads();
  for (int idx = tid; idx < 64 * Dn; idx += 256) {
    int row = idx >> 8, col = idx & 255;
    a_lds[idx] = (_Float16)emb[(size_t)toks[row] * Dn + col];
  }
  __syncthreads();

  const int wave = tid >> 5, lane = tid & 31;
  const _Float16* wih = dir ? wihB : wihF;
  const float* bih = dir ? bihB : bihF;
  const float* bhh = dir ? bhhB : bhhF;
  float* G = dir ? Gb : Gf;
  const int n0  = blockIdx.y * 128 + wave * 16;
  const int col = n0 + (lane & 15);
  const float bias = bih[col] + bhh[col];

#pragma unroll
  for (int mt = 0; mt < 4; ++mt) {
    v8f acc = {};
#pragma unroll
    for (int kc = 0; kc < 8; ++kc) {
      v16h a = load_frag16x32(a_lds, mt * 16, kc * 32, Dn, lane);
      v16h w = load_frag16x32(wih,   n0,      kc * 32, Dn, lane);
      acc = __builtin_amdgcn_wmma_f32_16x16x32_f16(false, a, false, w, (short)0, acc, false, false);
    }
    const int rb = mt * 16 + ((lane >> 4) << 3);
#pragma unroll
    for (int r = 0; r < 8; ++r)
      G[(size_t)(r0 + rb + r) * G4 + col] = acc[r] + bias;
  }
}

// ---------------------------------------------------------------------------
// Phase 2: persistent LSTM recurrence, one workgroup (32 waves) per direction.
// Per step: gates[64,1024] = h[64,256] @ whh^T via 2048 WMMAs, then fused
// activation; cell state c kept in registers (thread tid owns b=tid/16,
// j in [(tid%16)*16, +16)). Backward dir indexes gates/output at rev(b,t).
// Epilogue uses float4 / uint4 vector traffic and prefetches next step's
// precomputed gates (global_prefetch) to shorten the serial chain.
// Dynamic LDS: h (f16 32KB) + gates (f32 256KB) = 288KB.
// ---------------------------------------------------------------------------
__global__ void __launch_bounds__(1024)
lstm_recurrence_kernel(const float* __restrict__ Gf, const float* __restrict__ Gb,
                       const _Float16* __restrict__ whhF, const _Float16* __restrict__ whhB,
                       const int* __restrict__ lengths,
                       _Float16* __restrict__ hcat) {
  extern __shared__ char smem[];
  _Float16* h_lds = (_Float16*)smem;                                     // [64][256]
  float*    gates = (float*)(smem + (size_t)Bn * Hn * sizeof(_Float16)); // [64][1024]

  const int dir  = blockIdx.x;
  const int tid  = threadIdx.x;
  const int wave = tid >> 5;
  const int lane = tid & 31;
  const _Float16* whh = dir ? whhB : whhF;
  const float*    Gp  = dir ? Gb : Gf;

  const int b   = tid >> 4;
  const int j0  = (tid & 15) << 4;
  const int len = lengths[b];

  float c_reg[16];
#pragma unroll
  for (int i = 0; i < 16; ++i) c_reg[i] = 0.0f;
  for (int idx = tid; idx < Bn * Hn; idx += 1024) h_lds[idx] = (_Float16)0.0f;

  // warm L2 for the first step's precomputed gates
  {
    const int ts0 = (dir == 0) ? 0 : ((0 < len) ? (len - 1) : 0);
    const char* pn = (const char*)(Gp + (size_t)(b * Tn + ts0) * G4);
#pragma unroll
    for (int g = 0; g < 4; ++g) __builtin_prefetch(pn + (g * 256 + j0) * 4, 0, 1);
  }
  __syncthreads();

  for (int t = 0; t < Tn; ++t) {
    // --- recurrent GEMM: M=64, N=1024, K=256; wave owns 2 n-tiles ---
#pragma unroll
    for (int nn = 0; nn < 2; ++nn) {
      const int n0 = wave * 32 + nn * 16;
#pragma unroll
      for (int mt = 0; mt < 4; ++mt) {
        v8f acc = {};
#pragma unroll
        for (int kc = 0; kc < 8; ++kc) {
          v16h a = load_frag16x32(h_lds, mt * 16, kc * 32, Hn, lane);
          v16h w = load_frag16x32(whh,   n0,      kc * 32, Hn, lane);
          acc = __builtin_amdgcn_wmma_f32_16x16x32_f16(false, a, false, w, (short)0, acc, false, false);
        }
        const int col = n0 + (lane & 15);
        const int rb  = mt * 16 + ((lane >> 4) << 3);
#pragma unroll
        for (int r = 0; r < 8; ++r) gates[(rb + r) * G4 + col] = acc[r];
      }
    }
    __syncthreads();

    // --- fused LSTM cell (PyTorch gate order i,f,g,o) ---
    const int tsrc = (dir == 0) ? t : ((t < len) ? (len - 1 - t) : t);
    const float* gp = Gp + (size_t)(b * Tn + tsrc) * G4;
    _Float16* hout = hcat + (size_t)(b * Tn + tsrc) * (2 * Hn) + dir * Hn;

    // prefetch next step's precomputed gate rows while this step computes
    if (t + 1 < Tn) {
      const int tn   = t + 1;
      const int tsn  = (dir == 0) ? tn : ((tn < len) ? (len - 1 - tn) : tn);
      const char* pn = (const char*)(Gp + (size_t)(b * Tn + tsn) * G4);
#pragma unroll
      for (int g = 0; g < 4; ++g) __builtin_prefetch(pn + (g * 256 + j0) * 4, 0, 1);
    }

    float iv[16], fv[16], gv[16], ov[16];
    float pi[16], pf[16], pg[16], po[16];
    const float* gl = gates + b * G4;
    load16f(iv, gl + j0);         load16f(pi, gp + j0);
    load16f(fv, gl + 256 + j0);   load16f(pf, gp + 256 + j0);
    load16f(gv, gl + 512 + j0);   load16f(pg, gp + 512 + j0);
    load16f(ov, gl + 768 + j0);   load16f(po, gp + 768 + j0);

    union { _Float16 h[16]; uint4 q[2]; } hb;
#pragma unroll
    for (int i = 0; i < 16; ++i) {
      const float ig = sigf(iv[i] + pi[i]);
      const float fg = sigf(fv[i] + pf[i]);
      const float gg = tanhf(gv[i] + pg[i]);
      const float og = sigf(ov[i] + po[i]);
      const float c  = fg * c_reg[i] + ig * gg;
      c_reg[i] = c;
      hb.h[i] = (_Float16)(og * tanhf(c));
    }
    ((uint4*)(h_lds + b * Hn + j0))[0] = hb.q[0];
    ((uint4*)(h_lds + b * Hn + j0))[1] = hb.q[1];
    ((uint4*)(hout + j0))[0] = hb.q[0];
    ((uint4*)(hout + j0))[1] = hb.q[1];
    __syncthreads();
  }
}

// ---------------------------------------------------------------------------
// Phase 3: emission projection  em[r,c] = hcat[r,:] . w_out[c,:] + b_out[c]
//   R=16384, K=512, N=32 -> 2 n-tiles, 16 k-chunks per 16-row tile.
// ---------------------------------------------------------------------------
__global__ void __launch_bounds__(256)
emission_kernel(const _Float16* __restrict__ hcat, const _Float16* __restrict__ wout,
                const float* __restrict__ bout, float* __restrict__ em) {
  const int tid = threadIdx.x, wave = tid >> 5, lane = tid & 31;
  const int r0 = (blockIdx.x * 8 + wave) * 16;
#pragma unroll
  for (int nt = 0; nt < 2; ++nt) {
    const int n0 = nt * 16;
    v8f acc = {};
#pragma unroll
    for (int kc = 0; kc < 16; ++kc) {
      v16h a = load_frag16x32(hcat, r0, kc * 32, 2 * Hn, lane);
      v16h w = load_frag16x32(wout, n0, kc * 32, 2 * Hn, lane);
      acc = __builtin_amdgcn_wmma_f32_16x16x32_f16(false, a, false, w, (short)0, acc, false, false);
    }
    const int col = n0 + (lane & 15);
    const float bias = bout[col];
    const int rb = r0 + ((lane >> 4) << 3);
#pragma unroll
    for (int r = 0; r < 8; ++r) em[(size_t)(rb + r) * Cn + col] = acc[r] + bias;
  }
}

// ---------------------------------------------------------------------------
// Phase 4: Viterbi. One wave per batch element, lane = tag (C==32==wave32).
// The lane's transition column trans[:,lane] is hoisted into registers.
// Matches reference: hist recorded every step, score update masked by t<len,
// first-max tie-breaking, backtrack with t < len-1 condition.
// Output layout: d_out[0 : B*T) = paths (as float), d_out[B*T : B*T+B) = score.
// ---------------------------------------------------------------------------
__global__ void __launch_bounds__(1024)
viterbi_kernel(const float* __restrict__ em, const int* __restrict__ lengths,
               const float* __restrict__ start_trans, const float* __restrict__ end_trans,
               const float* __restrict__ trans, int* __restrict__ hist,
               float* __restrict__ out) {
  const int tid  = threadIdx.x;
  const int wave = tid >> 5;
  const int lane = tid & 31;             // current tag c
  const int b    = blockIdx.x * 32 + wave;
  const int len  = lengths[b];

  // lane c caches trans[cp][c] for all cp
  float trow[Cn];
#pragma unroll
  for (int cp = 0; cp < Cn; ++cp) trow[cp] = trans[cp * Cn + lane];

  float score = start_trans[lane] + em[(size_t)(b * Tn) * Cn + lane];
  for (int t = 1; t < Tn; ++t) {
    float best = -3.4e38f;
    int   barg = 0;
#pragma unroll
    for (int cp = 0; cp < Cn; ++cp) {
      float v = __shfl(score, cp, 32) + trow[cp];
      if (v > best) { best = v; barg = cp; }
    }
    hist[(size_t)(t - 1) * (Bn * Cn) + b * Cn + lane] = barg;
    if (t < len) score = best + em[(size_t)(b * Tn + t) * Cn + lane];
  }
  score += end_trans[lane];

  float bestv = -3.4e38f;
  int   bestc = 0;
  for (int cc = 0; cc < Cn; ++cc) {
    float v = __shfl(score, cc, 32);
    if (v > bestv) { bestv = v; bestc = cc; }
  }
  if (lane == 0) {
    out[(size_t)Bn * Tn + b] = bestv;
    int tag = bestc;
    out[(size_t)b * Tn + (Tn - 1)] = (float)tag;
    for (int t = Tn - 2; t >= 0; --t) {
      if (t < len - 1) tag = hist[(size_t)t * (Bn * Cn) + b * Cn + tag];
      out[(size_t)b * Tn + t] = (float)tag;
    }
  }
}

// ---------------------------------------------------------------------------
extern "C" void kernel_launch(void* const* d_in, const int* in_sizes, int n_in,
                              void* d_out, int out_size, void* d_ws, size_t ws_size,
                              hipStream_t stream) {
  (void)in_sizes; (void)n_in; (void)out_size; (void)ws_size;
  const int*   sent    = (const int*)d_in[0];
  const int*   lengths = (const int*)d_in[1];
  const float* emb     = (const float*)d_in[2];
  const float* w_ih_f  = (const float*)d_in[3];
  const float* w_hh_f  = (const float*)d_in[4];
  const float* b_ih_f  = (const float*)d_in[5];
  const float* b_hh_f  = (const float*)d_in[6];
  const float* w_ih_b  = (const float*)d_in[7];
  const float* w_hh_b  = (const float*)d_in[8];
  const float* b_ih_b  = (const float*)d_in[9];
  const float* b_hh_b  = (const float*)d_in[10];
  const float* w_out   = (const float*)d_in[11];
  const float* b_out   = (const float*)d_in[12];
  const float* start_t = (const float*)d_in[13];
  const float* end_t   = (const float*)d_in[14];
  const float* trans   = (const float*)d_in[15];

  char* ws = (char*)d_ws;
  size_t off = 0;
  auto alloc = [&](size_t bytes) -> void* {
    void* p = ws + off;
    off += (bytes + 255) & ~(size_t)255;
    return p;
  };
  _Float16* wihF  = (_Float16*)alloc((size_t)G4 * Dn * sizeof(_Float16));
  _Float16* wihB  = (_Float16*)alloc((size_t)G4 * Dn * sizeof(_Float16));
  _Float16* whhF  = (_Float16*)alloc((size_t)G4 * Hn * sizeof(_Float16));
  _Float16* whhB  = (_Float16*)alloc((size_t)G4 * Hn * sizeof(_Float16));
  _Float16* woutH = (_Float16*)alloc((size_t)Cn * 2 * Hn * sizeof(_Float16));
  float*    Gf    = (float*)alloc((size_t)RT * G4 * sizeof(float));
  float*    Gb    = (float*)alloc((size_t)RT * G4 * sizeof(float));
  _Float16* hcat  = (_Float16*)alloc((size_t)RT * 2 * Hn * sizeof(_Float16));
  float*    emis  = (float*)alloc((size_t)RT * Cn * sizeof(float));
  int*      hist  = (int*)alloc((size_t)(Tn - 1) * Bn * Cn * sizeof(int));

  // Phase 0: weight conversion to f16
  const int nW = G4 * Dn;
  cvt_f16_kernel<<<(nW + 255) / 256, 256, 0, stream>>>(w_ih_f, wihF, nW);
  cvt_f16_kernel<<<(nW + 255) / 256, 256, 0, stream>>>(w_ih_b, wihB, nW);
  cvt_f16_kernel<<<(nW + 255) / 256, 256, 0, stream>>>(w_hh_f, whhF, nW);
  cvt_f16_kernel<<<(nW + 255) / 256, 256, 0, stream>>>(w_hh_b, whhB, nW);
  const int nO = Cn * 2 * Hn;
  cvt_f16_kernel<<<(nO + 255) / 256, 256, 0, stream>>>(w_out, woutH, nO);

  // Phase 1: embedding gather + input gates (both directions)
  input_gates_kernel<<<dim3(RT / 64, G4 / 128, 2), 256, 0, stream>>>(
      sent, emb, wihF, wihB, b_ih_f, b_hh_f, b_ih_b, b_hh_b, Gf, Gb);

  // Phase 2: persistent BiLSTM recurrence (1 WGP per direction)
  const size_t lds2 = (size_t)Bn * Hn * sizeof(_Float16) + (size_t)Bn * G4 * sizeof(float);
  hipFuncSetAttribute((const void*)lstm_recurrence_kernel,
                      hipFuncAttributeMaxDynamicSharedMemorySize, (int)lds2);
  lstm_recurrence_kernel<<<2, 1024, lds2, stream>>>(Gf, Gb, whhF, whhB, lengths, hcat);

  // Phase 3: emission projection
  emission_kernel<<<RT / 16 / 8, 256, 0, stream>>>(hcat, woutH, b_out, emis);

  // Phase 4: Viterbi decode + backtrack
  viterbi_kernel<<<2, 1024, 0, stream>>>(emis, lengths, start_t, end_t, trans,
                                         hist, (float*)d_out);
}
